// ForwardTransformLayer_15324443312357
// MI455X (gfx1250) — matrix operand order repaired and checked by reading, not compiled, not run on hardware
//
#include <hip/hip_runtime.h>
#include <cstdint>
#include <cstddef>

typedef __attribute__((ext_vector_type(16))) _Float16 v16h;
typedef __attribute__((ext_vector_type(8)))  float    v8f;

namespace {
constexpr int kH    = 4096;                 // per-half row length (even / odd)
constexpr int kCin  = 8192;                 // input row length (interleaved)
constexpr int kRows = 4096;
constexpr int kS    = 256;                  // output columns per workgroup per half
constexpr int kRS   = 532;                  // LDS row stride in floats (2*kS+16 = 528, padded)
constexpr int kChunksPerRow = (2 * kS + 16) / 4;   // 132 float4 chunks per row
constexpr int kNChunk = 16 * kChunksPerRow;        // 2112 chunks per workgroup
}

// CDNA5 async global -> LDS copy (16 bytes). Tracked by ASYNCcnt.
__device__ __forceinline__ void async_copy16(uint32_t lds_byte, const float* g) {
  asm volatile("global_load_async_to_lds_b128 %0, %1, off"
               :: "v"(lds_byte), "v"(g) : "memory");
}

__device__ __forceinline__ v8f wmma_fp16(v16h a, v16h b, v8f c) {
  // 8 args: (neg_a, A, neg_b, B, c_mod, C, reuse_a, reuse_b)
  return __builtin_amdgcn_wmma_f32_16x16x32_f16(false, a, false, b, (short)0, c,
                                                false, false);
}

__global__ __launch_bounds__(256) void wavelet_lift_kernel(
    const float* __restrict__ inp,      // (4096, 8192)
    const float* __restrict__ scaling,  // (8,)
    const float* __restrict__ srec,     // (8,)
    float* __restrict__ out)            // even_out (16M) then odd_out (16M)
{
  // 16 rows x 528 interleaved floats (8-pair circular halo on the left), padded
  // stride to stagger banks; +16 floats at the end hold the two filters.
  __shared__ __align__(16) float sm[16 * kRS + 16];
  float* fsm = &sm[16 * kRS];            // fsm[0..7]=wavelet, fsm[8..15]=scaling

  const int tid  = threadIdx.x;
  const int lane = tid & 31;             // wave32
  const int wave = tid >> 5;             // 8 waves / workgroup
  const int k0   = kS * blockIdx.x;      // output-column base (per half)
  const int row0 = 16 * blockIdx.y;      // row base

  float* outE = out;
  float* outO = out + (size_t)kRows * kH;

  // ---- Stage phase: async-copy the interleaved window [2*k0-16, 2*(k0+kS))
  // (mod 8192, parity preserved -> circular conv wrap handled here).
  for (int q = tid; q < kNChunk; q += 256) {
    int rr = q / kChunksPerRow;
    int cc = q - rr * kChunksPerRow;
    int g  = 2 * k0 - 16 + 4 * cc;
    if (g < 0) g += kCin;                                  // left wrap only
    const float* gp = inp + (size_t)(row0 + rr) * kCin + g;
    uint32_t loff = (uint32_t)(uintptr_t)&sm[rr * kRS + 4 * cc];
    async_copy16(loff, gp);
  }
  // Filters into LDS (dynamic-index reads later without register spills).
  if (tid < 8) {
    fsm[tid]     = srec[7 - tid] * ((tid & 1) ? -1.0f : 1.0f);  // wavelet
    fsm[8 + tid] = scaling[tid];
  }
  asm volatile("s_wait_asynccnt 0" ::: "memory");
  __syncthreads();

  // ---- Banded-weight B fragments (32x16 f16, f32 accuracy via hi/lo split).
  // B[p][n] = f[n+7-p] for n <= p <= n+7 else 0, so
  //   y[n] = sum_{p} x[c-7+p] * B[p][n] = sum_j f[j] * x[c+n-j].
  // Layout chosen here (contiguous K, same as A): VGPR v holds K=2v,2v+1; lanes16-31 hold K+16.
  const int n     = lane & 15;
  const int pbase = (lane < 16) ? 0 : 16;
  v16h bWh{}, bWl{}, bSh{}, bSl{};
#pragma unroll
  for (int e = 0; e < 16; ++e) {
    int p = pbase + e;
    int j = n + 7 - p;
    bool inband = (j >= 0) && (j < 8);
    int jc = inband ? j : 0;
    float fw = fsm[jc];
    float fs = fsm[8 + jc];
    if (!inband) { fw = 0.0f; fs = 0.0f; }
    _Float16 h = (_Float16)fw; bWh[e] = h; bWl[e] = (_Float16)(fw - (float)h);
    h = (_Float16)fs;          bSh[e] = h; bSl[e] = (_Float16)(fs - (float)h);
  }

  // ---- Compute: each wave does 2 column tiles of 16; 6 WMMAs per tile.
  const int m  = lane & 15;                  // A: row within tile
  const int kb = (lane < 16) ? 0 : 16;       // A lane-split of K
  const float* rowp = &sm[m * kRS];
  const int mtop = (lane < 16) ? 0 : 8;      // C/D: M = v + mtop, N = lane&15

  for (int t = wave; t < kS / 16; t += 8) {
    const int cl = 16 * t;                   // tile base (local column)

    // A fragments (16x32): A[m][K] = x[m][cl + K - 7], K<=22 valid, rest 0.
    v16h aeh{}, ael{}, aoh{}, aol{};
#pragma unroll
    for (int e = 0; e < 16; ++e) {
      int K  = kb + e;                       // per-lane K slots (contiguous, matches B)
      int Kc = (K <= 22) ? K : 0;            // clamp for a safe LDS address
      int base = 2 * (cl + Kc - 7) + 16;     // interleaved index (even at +0)
      float xe = rowp[base + 0];
      float xo = rowp[base + 1];
      if (K > 22) { xe = 0.0f; xo = 0.0f; }  // finite zeros: no 0*Inf NaNs
      _Float16 h = (_Float16)xe; aeh[e] = h; ael[e] = (_Float16)(xe - (float)h);
      h = (_Float16)xo;          aoh[e] = h; aol[e] = (_Float16)(xo - (float)h);
    }

    v8f accW = {};                           // conv(even, wavelet)
    accW = wmma_fp16(ael, bWh, accW);
    accW = wmma_fp16(aeh, bWl, accW);
    accW = wmma_fp16(aeh, bWh, accW);

    v8f accS = {};                           // conv(odd, scaling)
    accS = wmma_fp16(aol, bSh, accS);
    accS = wmma_fp16(aoh, bSl, accS);
    accS = wmma_fp16(aoh, bSh, accS);

    // D layout: lane holds N = lane&15, M = v + 8*(lane/16) in element v.
#pragma unroll
    for (int v = 0; v < 8; ++v) {
      int mm = mtop + v;
      int basec = mm * kRS + 2 * (cl + n) + 16;
      float ce = sm[basec + 0];
      float co = sm[basec + 1];
      size_t off = (size_t)(row0 + mm) * kH + (size_t)(k0 + cl + n);
      outE[off] = ce - accS[v];              // even - conv(odd, scaling)
      outO[off] = co - accW[v];              // odd  - conv(even, wavelet)
    }
  }
}

extern "C" void kernel_launch(void* const* d_in, const int* in_sizes, int n_in,
                              void* d_out, int out_size, void* d_ws, size_t ws_size,
                              hipStream_t stream) {
  (void)in_sizes; (void)n_in; (void)out_size; (void)d_ws; (void)ws_size;
  const float* inp     = (const float*)d_in[0];
  const float* scaling = (const float*)d_in[1];
  const float* srec    = (const float*)d_in[2];
  float* out = (float*)d_out;

  dim3 grid(kH / kS, kRows / 16);   // (16, 256) workgroups
  dim3 block(256);                  // 8 wave32 waves
  wavelet_lift_kernel<<<grid, block, 0, stream>>>(inp, scaling, srec, out);
}
